// HexTransformerLayer_72172630442352
// MI455X (gfx1250) — matrix operand order repaired and checked by reading, not compiled, not run on hardware
//
#include <hip/hip_runtime.h>
#include <math.h>

typedef __bf16 bf16_t;
typedef __attribute__((ext_vector_type(16))) __bf16 v16bf;
typedef __attribute__((ext_vector_type(8)))  __bf16 v8bf;
typedef __attribute__((ext_vector_type(8)))  float  v8f;
typedef __attribute__((ext_vector_type(4))) unsigned int u32x4;
typedef __attribute__((ext_vector_type(8))) int          i32x8;
typedef __attribute__((ext_vector_type(4))) int          i32x4;

#define D_MODEL 256
#define NUM_HEADS 8
#define D_HEAD 32
#define D_FF 1024
#define BB 4
#define NN 2048
#define MROWS (BB * NN)   // 8192 token rows

static __device__ __forceinline__ v16bf join8(v8bf lo, v8bf hi) {
  return __builtin_shufflevector(lo, hi, 0,1,2,3,4,5,6,7,8,9,10,11,12,13,14,15);
}

static __device__ __forceinline__ v8f wmma_bf16(v16bf a, v16bf b, v8f c) {
  // D = A(16x32) * B(32x16) + C, fp32 accumulate
  return __builtin_amdgcn_wmma_f32_16x16x32_bf16(false, a, false, b, (short)0, c, false, false);
}

// ---------------------------------------------------------------------------
// TDM: DMA a 32x32 bf16 tile (row stride `stride_elems`) from global into LDS.
// D# packing per CDNA5 ISA ch.8 (group0: count/lds_addr/global_addr/type=2;
// group1: data_size=2B, tensor_dim0=32, tensor_dim1=32, tile 32x32, stride0).
// Issue from one wave only; tracked by TENSORcnt.
// Toolchain here exposes the 6-arg builtin (clang-23 / therock-10.0 form).
// ---------------------------------------------------------------------------
static __device__ __forceinline__ void tdm_load_tile32x32_bf16(
    unsigned int lds_off, const bf16_t* gptr, unsigned int stride_elems) {
  unsigned long long ga = (unsigned long long)(uintptr_t)gptr;
  u32x4 g0;
  g0[0] = 1u;                                   // count=1 (valid user D#)
  g0[1] = lds_off;                              // lds_addr (bytes)
  g0[2] = (unsigned int)ga;                     // global_addr[31:0]
  g0[3] = (unsigned int)((ga >> 32) & 0x1FFFFFFu) | (2u << 30);  // addr hi, type=2
  i32x8 g1;
  g1[0] = (1 << 16);                            // wg_mask=0, data_size=1 (2 bytes)
  g1[1] = (32 << 16);                           // tensor_dim0=32 (low 16 bits)
  g1[2] = (32 << 16);                           // tensor_dim0 hi=0 | tensor_dim1=32 lo
  g1[3] = (32 << 16);                           // tensor_dim1 hi=0 | tile_dim0=32
  g1[4] = 32;                                   // tile_dim1=32, tile_dim2=0
  g1[5] = (int)stride_elems;                    // tensor_dim0_stride lo32
  g1[6] = 0;                                    // stride0 hi | stride1 lo
  g1[7] = 0;                                    // stride1 hi
  i32x4 gz4 = {0, 0, 0, 0};                     // groups 2/3 unused (2D tensor)
  i32x8 gz8 = {0, 0, 0, 0, 0, 0, 0, 0};         // trailing group (zero-filled)
  __builtin_amdgcn_tensor_load_to_lds(g0, g1, gz4, gz4, gz8, 0);
}

// ---------------------------------------------------------------------------
// Weight transpose + fp32 -> bf16 convert:  src[K, Nc] row-major -> dst[Nc, K]
// ---------------------------------------------------------------------------
__global__ void transpose_to_bf16_kernel(const float* __restrict__ src,
                                         bf16_t* __restrict__ dst,
                                         int K, int Nc) {
  int idx = blockIdx.x * blockDim.x + threadIdx.x;
  int total = K * Nc;
  if (idx >= total) return;
  int k = idx / Nc;
  int n = idx - k * Nc;
  dst[(size_t)n * K + k] = (bf16_t)src[idx];
}

// ---------------------------------------------------------------------------
// LayerNorm (one wave per 256-wide row) -> bf16 output
// ---------------------------------------------------------------------------
__global__ __launch_bounds__(128) void layernorm_bf16_kernel(
    const float* __restrict__ x, const float* __restrict__ g,
    const float* __restrict__ b, bf16_t* __restrict__ out, int M) {
  const int lane = threadIdx.x & 31;
  const int wave = threadIdx.x >> 5;
  const int row  = blockIdx.x * 4 + wave;
  if (row >= M) return;
  const float* xr = x + (size_t)row * D_MODEL;
  float vals[8];
  float s = 0.f;
#pragma unroll
  for (int i = 0; i < 8; ++i) { vals[i] = xr[lane * 8 + i]; s += vals[i]; }
#pragma unroll
  for (int off = 16; off > 0; off >>= 1) s += __shfl_xor(s, off, 32);
  const float mu = s * (1.0f / D_MODEL);
  float q = 0.f;
#pragma unroll
  for (int i = 0; i < 8; ++i) { float d = vals[i] - mu; q += d * d; }
#pragma unroll
  for (int off = 16; off > 0; off >>= 1) q += __shfl_xor(q, off, 32);
  const float rstd = rsqrtf(q * (1.0f / D_MODEL) + 1e-5f);
  bf16_t* orow = out + (size_t)row * D_MODEL;
#pragma unroll
  for (int i = 0; i < 8; ++i) {
    int c = lane * 8 + i;
    orow[c] = (bf16_t)((vals[i] - mu) * rstd * g[c] + b[c]);
  }
}

// ---------------------------------------------------------------------------
// Generic WMMA GEMM:  out[M,Nc] = epilogue( A[M,K](bf16) * Wt[Nc,K]^T + bias )
// Block = 4 waves; block tile 64x64; wave tile 16x64 (4 accumulators).
// mode 0: bf16 out            (QKV projection)
// mode 1: f32 out + residual  (out-proj, FF2)
// mode 2: exact GELU, bf16 out (FF1)
// ---------------------------------------------------------------------------
__global__ __launch_bounds__(128) void gemm_wmma_kernel(
    const bf16_t* __restrict__ A, const bf16_t* __restrict__ Wt,
    const float* __restrict__ bias, const float* __restrict__ res,
    void* __restrict__ outp, int M, int Nc, int K, int mode) {
  const int lane = threadIdx.x & 31;
  const int wave = threadIdx.x >> 5;
  const int hl = lane >> 4;     // half-wave select
  const int ll = lane & 15;
  const int m0 = blockIdx.y * 64 + wave * 16;
  const int n0 = blockIdx.x * 64;

  v8f zero = {};
  v8f acc[4] = {zero, zero, zero, zero};

  const bf16_t* arow = A + (size_t)(m0 + ll) * K;
  for (int kk = 0; kk < K; kk += 32) {
    const int akb = kk + hl * 16;
    v8bf a_lo = *(const v8bf*)(arow + akb);
    v8bf a_hi = *(const v8bf*)(arow + akb + 8);
    v16bf afrag = join8(a_lo, a_hi);
    const int bkb = kk + hl * 16;
#pragma unroll
    for (int c = 0; c < 4; ++c) {
      const int col = n0 + c * 16 + ll;
      v16bf bfrag = *(const v16bf*)(Wt + (size_t)col * K + bkb);
      acc[c] = wmma_bf16(afrag, bfrag, acc[c]);
    }
  }

#pragma unroll
  for (int c = 0; c < 4; ++c) {
#pragma unroll
    for (int r = 0; r < 8; ++r) {
      const int row = m0 + r + hl * 8;
      const int col = n0 + c * 16 + ll;
      const size_t idx = (size_t)row * Nc + col;
      float v = acc[c][r] + bias[col];
      if (mode == 0) {
        ((bf16_t*)outp)[idx] = (bf16_t)v;
      } else if (mode == 1) {
        ((float*)outp)[idx] = v + res[idx];
      } else {
        float gl = 0.5f * v * (1.0f + erff(v * 0.70710678118654752f));
        ((bf16_t*)outp)[idx] = (bf16_t)gl;
      }
    }
  }
}

// ---------------------------------------------------------------------------
// Flash attention with additive structural bias.
// qkv: bf16 [B*N, 768] columns = {q,k,v} x head x dh.
// Block = 4 waves sharing one (b,h); per 32-key chunk the K and V tiles are
// DMA'd into LDS once via the Tensor Data Mover (tensor_load_to_lds,
// TENSORcnt), V is transposed cooperatively in LDS, then each wave runs
// 16-query WMMA logits + online softmax + WMMA PV.
// ---------------------------------------------------------------------------
__global__ __launch_bounds__(128) void attention_kernel(
    const bf16_t* __restrict__ qkv, const float* __restrict__ bias,
    bf16_t* __restrict__ attn_out) {
  __shared__ __align__(32) bf16_t lds_k[32 * 32];    // [key][dh] via TDM
  __shared__ __align__(32) bf16_t lds_v[32 * 32];    // [key][dh] via TDM
  __shared__ __align__(32) bf16_t lds_vt[32 * 32];   // [dh][key]
  __shared__ __align__(32) bf16_t lds_p[4][16 * 32]; // per-wave P tile

  const int lane = threadIdx.x & 31;
  const int wave = threadIdx.x >> 5;
  const int hl = lane >> 4;
  const int ll = lane & 15;
  const int b = blockIdx.y / NUM_HEADS;
  const int h = blockIdx.y - b * NUM_HEADS;
  const int q0 = blockIdx.x * 64 + wave * 16;

  const size_t rs = 3 * D_MODEL;  // 768 qkv row stride
  const bf16_t* qbase = qkv + (size_t)b * NN * rs + h * D_HEAD;
  const bf16_t* kg = qbase + D_MODEL;
  const bf16_t* vg = qbase + 2 * D_MODEL;
  const float* biash = bias + (size_t)h * NN * NN;

  const unsigned int koff = (unsigned int)(uintptr_t)&lds_k[0];
  const unsigned int voff = (unsigned int)(uintptr_t)&lds_v[0];

  // Q tile as A-fragment (M=16 queries, K=dh=32)
  v16bf qfrag;
  {
    const bf16_t* qr = qbase + (size_t)(q0 + ll) * rs + hl * 16;
    qfrag = join8(*(const v8bf*)(qr), *(const v8bf*)(qr + 8));
  }

  v8f zero = {};
  v8f o0 = zero, o1 = zero;
  float mrun[8], lrun[8];
#pragma unroll
  for (int r = 0; r < 8; ++r) { mrun[r] = -3.0e38f; lrun[r] = 0.f; }
  const float scale = 0.17677669529663687f;  // 1/sqrt(dh)

  bf16_t* pbuf = lds_p[wave];

  for (int kc = 0; kc < NN; kc += 32) {
    __syncthreads();  // previous chunk's consumers done
    if (wave == 0) {
      tdm_load_tile32x32_bf16(koff, kg + (size_t)kc * rs, (unsigned int)rs);
      tdm_load_tile32x32_bf16(voff, vg + (size_t)kc * rs, (unsigned int)rs);
      __builtin_amdgcn_s_wait_tensorcnt(0);
    }
    __syncthreads();  // K/V tiles resident in LDS

    // cooperative V transpose: lds_v[key][d] -> lds_vt[d][key]
    {
      const int tk = threadIdx.x & 31;
      const int td = (threadIdx.x >> 5) * 8;
      v8bf vv = *(const v8bf*)(lds_v + tk * 32 + td);
#pragma unroll
      for (int j = 0; j < 8; ++j) lds_vt[(td + j) * 32 + tk] = vv[j];
    }
    __syncthreads();

    // --- logits: two 16-key tiles via WMMA; B-fragments from LDS (contiguous)
    v16bf kf0 = *(const v16bf*)(lds_k + ll * 32 + hl * 16);
    v16bf kf1 = *(const v16bf*)(lds_k + (16 + ll) * 32 + hl * 16);
    v8f s0 = wmma_bf16(qfrag, kf0, zero);
    v8f s1 = wmma_bf16(qfrag, kf1, zero);

    // prefetch next bias chunk (speculative)
    {
      const float* pf = &biash[(size_t)(q0 + hl * 8) * NN + ((kc + 32) & (NN - 1)) + ll];
      __builtin_prefetch(pf, 0, 1);
    }

    // --- scale + structural bias + per-row max (rows live on 16-lane halves)
    float pm[8];
#pragma unroll
    for (int r = 0; r < 8; ++r) {
      const int qrow = q0 + r + hl * 8;
      const float b0 = biash[(size_t)qrow * NN + kc + ll];
      const float b1 = biash[(size_t)qrow * NN + kc + 16 + ll];
      s0[r] = s0[r] * scale + b0;
      s1[r] = s1[r] * scale + b1;
      float t = fmaxf(s0[r], s1[r]);
#pragma unroll
      for (int off = 1; off < 16; off <<= 1) t = fmaxf(t, __shfl_xor(t, off, 16));
      pm[r] = t;
    }

    // --- online softmax update; stash P (bf16) into LDS row-major [16][32]
#pragma unroll
    for (int r = 0; r < 8; ++r) {
      const float mnew = fmaxf(mrun[r], pm[r]);
      const float corr = __expf(mrun[r] - mnew);
      const float p0 = __expf(s0[r] - mnew);
      const float p1 = __expf(s1[r] - mnew);
      float rsum = p0 + p1;
#pragma unroll
      for (int off = 1; off < 16; off <<= 1) rsum += __shfl_xor(rsum, off, 16);
      lrun[r] = lrun[r] * corr + rsum;
      mrun[r] = mnew;
      o0[r] *= corr;
      o1[r] *= corr;
      const int prow = r + hl * 8;
      pbuf[prow * 32 + ll]      = (bf16_t)p0;
      pbuf[prow * 32 + 16 + ll] = (bf16_t)p1;
    }
    asm volatile("s_wait_dscnt 0x0" ::: "memory");  // LDS turnaround (same wave)

    // --- P as A-fragment, V^T columns as B-fragments; O += P * V
    v16bf pfrag;
    {
      const bf16_t* pr = pbuf + ll * 32 + hl * 16;
      pfrag = join8(*(const v8bf*)(pr), *(const v8bf*)(pr + 8));
    }
    v16bf vf0 = *(const v16bf*)(lds_vt + ll * 32 + hl * 16);
    v16bf vf1 = *(const v16bf*)(lds_vt + (16 + ll) * 32 + hl * 16);
    o0 = wmma_bf16(pfrag, vf0, o0);
    o1 = wmma_bf16(pfrag, vf1, o1);
  }

  // --- normalize and scatter to [B*N, D_MODEL] at head columns
  bf16_t* outb = attn_out + (size_t)b * NN * D_MODEL + h * D_HEAD;
#pragma unroll
  for (int r = 0; r < 8; ++r) {
    const int qrow = q0 + r + hl * 8;
    const float inv = 1.0f / lrun[r];
    outb[(size_t)qrow * D_MODEL + ll]      = (bf16_t)(o0[r] * inv);
    outb[(size_t)qrow * D_MODEL + 16 + ll] = (bf16_t)(o1[r] * inv);
  }
}

// ---------------------------------------------------------------------------
extern "C" void kernel_launch(void* const* d_in, const int* in_sizes, int n_in,
                              void* d_out, int out_size, void* d_ws, size_t ws_size,
                              hipStream_t stream) {
  (void)in_sizes; (void)n_in; (void)out_size; (void)ws_size;
  const float* x     = (const float*)d_in[0];
  const float* sbias = (const float*)d_in[1];
  const float* ln1_g = (const float*)d_in[2];
  const float* ln1_b = (const float*)d_in[3];
  const float* ln2_g = (const float*)d_in[4];
  const float* ln2_b = (const float*)d_in[5];
  const float* qkv_w = (const float*)d_in[6];
  const float* qkv_b = (const float*)d_in[7];
  const float* out_w = (const float*)d_in[8];
  const float* out_b = (const float*)d_in[9];
  const float* ff1_w = (const float*)d_in[10];
  const float* ff1_b = (const float*)d_in[11];
  const float* ff2_w = (const float*)d_in[12];
  const float* ff2_b = (const float*)d_in[13];

  char* ws = (char*)d_ws;
  size_t off = 0;
  auto alloc = [&](size_t bytes) {
    void* p = ws + off;
    off += (bytes + 255) & ~(size_t)255;
    return p;
  };

  bf16_t* qkv_wt  = (bf16_t*)alloc((size_t)768 * 256 * 2);
  bf16_t* out_wt  = (bf16_t*)alloc((size_t)256 * 256 * 2);
  bf16_t* ff1_wt  = (bf16_t*)alloc((size_t)D_FF * 256 * 2);
  bf16_t* ff2_wt  = (bf16_t*)alloc((size_t)256 * D_FF * 2);
  bf16_t* normed1 = (bf16_t*)alloc((size_t)MROWS * D_MODEL * 2);
  bf16_t* qkvbuf  = (bf16_t*)alloc((size_t)MROWS * 768 * 2);
  bf16_t* attnout = (bf16_t*)alloc((size_t)MROWS * D_MODEL * 2);
  float*  x1      = (float*) alloc((size_t)MROWS * D_MODEL * 4);
  bf16_t* normed2 = (bf16_t*)alloc((size_t)MROWS * D_MODEL * 2);
  bf16_t* ffa     = (bf16_t*)alloc((size_t)MROWS * D_FF * 2);

  // 1) weights -> bf16 transposed
  transpose_to_bf16_kernel<<<(256 * 768 + 255) / 256, 256, 0, stream>>>(qkv_w, qkv_wt, 256, 768);
  transpose_to_bf16_kernel<<<(256 * 256 + 255) / 256, 256, 0, stream>>>(out_w, out_wt, 256, 256);
  transpose_to_bf16_kernel<<<(256 * D_FF + 255) / 256, 256, 0, stream>>>(ff1_w, ff1_wt, 256, D_FF);
  transpose_to_bf16_kernel<<<(D_FF * 256 + 255) / 256, 256, 0, stream>>>(ff2_w, ff2_wt, D_FF, 256);

  // 2) LN1
  layernorm_bf16_kernel<<<MROWS / 4, 128, 0, stream>>>(x, ln1_g, ln1_b, normed1, MROWS);

  // 3) QKV projection (bf16 out)
  gemm_wmma_kernel<<<dim3(768 / 64, MROWS / 64), 128, 0, stream>>>(
      normed1, qkv_wt, qkv_b, nullptr, qkvbuf, MROWS, 768, 256, 0);

  // 4) flash attention with structural bias (TDM-staged K/V)
  attention_kernel<<<dim3(NN / 64, BB * NUM_HEADS), 128, 0, stream>>>(qkvbuf, sbias, attnout);

  // 5) output projection + residual (f32)
  gemm_wmma_kernel<<<dim3(D_MODEL / 64, MROWS / 64), 128, 0, stream>>>(
      attnout, out_wt, out_b, x, x1, MROWS, D_MODEL, 256, 1);

  // 6) LN2
  layernorm_bf16_kernel<<<MROWS / 4, 128, 0, stream>>>(x1, ln2_g, ln2_b, normed2, MROWS);

  // 7) FF1 + exact GELU (bf16 out)
  gemm_wmma_kernel<<<dim3(D_FF / 64, MROWS / 64), 128, 0, stream>>>(
      normed2, ff1_wt, ff1_b, nullptr, ffa, MROWS, D_FF, 256, 2);

  // 8) FF2 + residual -> final f32 output
  gemm_wmma_kernel<<<dim3(D_MODEL / 64, MROWS / 64), 128, 0, stream>>>(
      ffa, ff2_wt, ff2_b, x1, (float*)d_out, MROWS, D_MODEL, D_FF, 1);
}
